// ItemEmbeddingLayer_74217034875540
// MI455X (gfx1250) — matrix operand order repaired and checked by reading, not compiled, not run on hardware
//
#include <hip/hip_runtime.h>
#include <hip/hip_bf16.h>

typedef __attribute__((ext_vector_type(8)))  __bf16 bf16x8;
typedef __attribute__((ext_vector_type(16))) __bf16 bf16x16;
typedef __attribute__((ext_vector_type(8)))  float  f32x8;

// ---- WMMA fragment helpers -------------------------------------------------
// A-fragment (16x32 bf16): lane<16 holds M=lane, K in {k0..k0+7} u {k0+16..k0+23};
// lane>=16 holds M=lane-16, K in {k0+8..k0+15} u {k0+24..k0+31}. Caller bakes the
// half*8 offset into p, so: lo = p[0..7], hi = p[16..23].
__device__ __forceinline__ bf16x16 ld_a16(const __bf16* p) {
  bf16x8 lo = *(const bf16x8*)p;
  bf16x8 hi = *(const bf16x8*)(p + 16);
  bf16x16 r;
#pragma unroll
  for (int i = 0; i < 8; ++i) { r[i] = lo[i]; r[i + 8] = hi[i]; }
  return r;
}

// B-fragment (32x16 bf16): lane n<16 holds column n rows {k0..k0+7} u {k0+16..k0+23},
// lanes 16..31 hold rows {k0+8..k0+15} u {k0+24..k0+31}. With W stored transposed
// ([fan_out][fan_in]) caller bakes half*8 into p: lo = p[0..7], hi = p[16..23].
__device__ __forceinline__ bf16x16 ld_b16(const __bf16* p) {
  bf16x8 lo = *(const bf16x8*)p;
  bf16x8 hi = *(const bf16x8*)(p + 16);
  bf16x16 r;
#pragma unroll
  for (int i = 0; i < 8; ++i) { r[i] = lo[i]; r[i + 8] = hi[i]; }
  return r;
}

__device__ __forceinline__ f32x8 wmma_bf16(bf16x16 a, bf16x16 b, f32x8 c) {
  return __builtin_amdgcn_wmma_f32_16x16x32_bf16(false, a, false, b, (short)0, c,
                                                 false, false);
}

__device__ __forceinline__ bf16x8 cvt8(float4 x0, float4 x1) {
  bf16x8 h;
  h[0] = (__bf16)x0.x; h[1] = (__bf16)x0.y; h[2] = (__bf16)x0.z; h[3] = (__bf16)x0.w;
  h[4] = (__bf16)x1.x; h[5] = (__bf16)x1.y; h[6] = (__bf16)x1.z; h[7] = (__bf16)x1.w;
  return h;
}

__device__ __forceinline__ f32x8 splat8(float v) {
  f32x8 a;
#pragma unroll
  for (int j = 0; j < 8; ++j) a[j] = v;
  return a;
}

// ---- weight transpose + f32->bf16 ------------------------------------------
__global__ void wtrans_kernel(const float* __restrict__ W, __bf16* __restrict__ Wt,
                              int fin, int fout) {
  int idx = blockIdx.x * blockDim.x + threadIdx.x;
  if (idx >= fin * fout) return;
  int k = idx / fout, n = idx - k * fout;
  Wt[(size_t)n * fin + k] = (__bf16)W[idx];
}

// ---- segment-sum via native f32 atomics (1 wave per edge, float4 reads) ----
__global__ __launch_bounds__(256)
void scatter_add_kernel(const float* __restrict__ src, const int* __restrict__ dstIdx,
                        const int* __restrict__ srcIdx, float* __restrict__ dst, int E) {
  int gw = (blockIdx.x * 256 + threadIdx.x) >> 5;
  int lane = threadIdx.x & 31;
  if (gw >= E) return;
  const float4 v = ((const float4*)(src + (size_t)srcIdx[gw] * 128))[lane];
  float* d = dst + (size_t)dstIdx[gw] * 128 + lane * 4;
  unsafeAtomicAdd(d + 0, v.x);
  unsafeAtomicAdd(d + 1, v.y);
  unsafeAtomicAdd(d + 2, v.z);
  unsafeAtomicAdd(d + 3, v.w);
}

// ---- fused MLP2: relu(X@W1+b1)@W2+b2, X rows optionally gathered -----------
// 256 threads = 8 waves; 64-row tile. Each wave owns output-column tiles (ni)
// and keeps all K-step B fragments (weights) in VGPRs while sweeping the row
// tiles (mi), so each weight byte is read from L2 exactly once per workgroup.
__global__ __launch_bounds__(256)
void mlp2_kernel(const float* __restrict__ X, const int* __restrict__ gidx, int nRows,
                 const __bf16* __restrict__ W1t, const float* __restrict__ b1,
                 const __bf16* __restrict__ W2t, const float* __restrict__ b2,
                 float* __restrict__ out, int outStride, int colOff) {
  __shared__ __bf16 sX[64 * 128];   // 16 KB
  __shared__ __bf16 sH[64 * 256];   // 32 KB
  const int tid = threadIdx.x;
  const int m0g = blockIdx.x * 64;

  { // stage input tile fp32 -> bf16
    int r = tid >> 2;
    int cb = (tid & 3) * 32;
    int gr = m0g + r;
    bool ok = gr < nRows;
    long src = ok ? (gidx ? (long)gidx[gr] : (long)gr) : 0;
    const float4* p = (const float4*)(X + src * 128 + cb);
#pragma unroll
    for (int i = 0; i < 4; ++i) {
      bf16x8 h;
      if (ok) h = cvt8(p[2 * i], p[2 * i + 1]);
      else {
#pragma unroll
        for (int j = 0; j < 8; ++j) h[j] = (__bf16)0.0f;
      }
      *(bf16x8*)&sX[r * 128 + cb + i * 8] = h;
    }
  }
  __syncthreads();

  const int lane = tid & 31;
  const int wave = tid >> 5;
  const int nl = lane & 15;
  const int half = lane >> 4;

  // layer 1: [64x128]@[128x256]+b1, relu -> sH   (16 ni tiles, 2 per wave)
#pragma unroll 1
  for (int ni = wave; ni < 16; ni += 8) {
    const __bf16* bp = W1t + (size_t)(ni * 16 + nl) * 128 + half * 8;
    bf16x16 b[4];
#pragma unroll
    for (int kk = 0; kk < 4; ++kk) b[kk] = ld_b16(bp + kk * 32);
    float bias = b1[ni * 16 + nl];
    int col = ni * 16 + nl;
#pragma unroll
    for (int mi = 0; mi < 4; ++mi) {
      f32x8 acc = splat8(bias);
      const __bf16* ap = sX + (mi * 16 + nl) * 128 + half * 8;
#pragma unroll
      for (int kk = 0; kk < 4; ++kk)
        acc = wmma_bf16(ld_a16(ap + kk * 32), b[kk], acc);
#pragma unroll
      for (int j = 0; j < 8; ++j) {
        float v = acc[j] > 0.0f ? acc[j] : 0.0f;
        sH[(mi * 16 + half * 8 + j) * 256 + col] = (__bf16)v;
      }
    }
  }
  __syncthreads();

  // layer 2: [64x256]@[256x128]+b2 -> out column stripe (8 ni tiles, 1/wave)
  {
    int ni = wave;
    const __bf16* bp = W2t + (size_t)(ni * 16 + nl) * 256 + half * 8;
    bf16x16 b[8];
#pragma unroll
    for (int kk = 0; kk < 8; ++kk) b[kk] = ld_b16(bp + kk * 32);
    float bias = b2[ni * 16 + nl];
    int col = colOff + ni * 16 + nl;
#pragma unroll 1
    for (int mi = 0; mi < 4; ++mi) {
      f32x8 acc = splat8(bias);
      const __bf16* ap = sH + (mi * 16 + nl) * 256 + half * 8;
#pragma unroll
      for (int kk = 0; kk < 8; ++kk)
        acc = wmma_bf16(ld_a16(ap + kk * 32), b[kk], acc);
      int gr0 = m0g + mi * 16 + half * 8;
#pragma unroll
      for (int j = 0; j < 8; ++j) {
        int gr = gr0 + j;
        if (gr < nRows) out[(size_t)gr * outStride + col] = acc[j];
      }
    }
  }
}

// ---- fused MLP3 on concat [Nx512]: 512->256 relu ->128 relu ->128 ----------
__global__ __launch_bounds__(256)
void mlp3_kernel(const float* __restrict__ X, int nRows,
                 const __bf16* __restrict__ W1t, const float* __restrict__ b1,
                 const __bf16* __restrict__ W2t, const float* __restrict__ b2,
                 const __bf16* __restrict__ W3t, const float* __restrict__ b3,
                 float* __restrict__ out) {
  __shared__ __bf16 sX[32 * 512];   // 32 KB
  __shared__ __bf16 sH1[32 * 256];  // 16 KB
  __shared__ __bf16 sH2[32 * 128];  //  8 KB
  const int tid = threadIdx.x;
  const int m0g = blockIdx.x * 32;

  { // stage 32x512 tile fp32 -> bf16
    int r = tid >> 3;
    int cb = (tid & 7) * 64;
    int gr = m0g + r;
    bool ok = gr < nRows;
    const float4* p = (const float4*)(X + (size_t)(ok ? gr : 0) * 512 + cb);
#pragma unroll
    for (int i = 0; i < 8; ++i) {
      bf16x8 h;
      if (ok) h = cvt8(p[2 * i], p[2 * i + 1]);
      else {
#pragma unroll
        for (int j = 0; j < 8; ++j) h[j] = (__bf16)0.0f;
      }
      *(bf16x8*)&sX[r * 512 + cb + i * 8] = h;
    }
  }
  __syncthreads();

  const int lane = tid & 31;
  const int wave = tid >> 5;
  const int nl = lane & 15;
  const int half = lane >> 4;

  // layer 1: [32x512]@[512x256]+b1, relu -> sH1  (16 ni tiles, 2 per wave;
  // K=512 split into two halves of 8 B-frags; both mi accumulators live)
#pragma unroll 1
  for (int ni = wave; ni < 16; ni += 8) {
    float bias = b1[ni * 16 + nl];
    f32x8 acc[2];
    acc[0] = splat8(bias);
    acc[1] = splat8(bias);
#pragma unroll
    for (int kh = 0; kh < 2; ++kh) {
      const __bf16* bp = W1t + (size_t)(ni * 16 + nl) * 512 + kh * 256 + half * 8;
      bf16x16 b[8];
#pragma unroll
      for (int kk = 0; kk < 8; ++kk) b[kk] = ld_b16(bp + kk * 32);
#pragma unroll
      for (int mi = 0; mi < 2; ++mi) {
        const __bf16* ap = sX + (mi * 16 + nl) * 512 + kh * 256 + half * 8;
#pragma unroll
        for (int kk = 0; kk < 8; ++kk)
          acc[mi] = wmma_bf16(ld_a16(ap + kk * 32), b[kk], acc[mi]);
      }
    }
    int col = ni * 16 + nl;
#pragma unroll
    for (int mi = 0; mi < 2; ++mi) {
#pragma unroll
      for (int j = 0; j < 8; ++j) {
        float v = acc[mi][j] > 0.0f ? acc[mi][j] : 0.0f;
        sH1[(mi * 16 + half * 8 + j) * 256 + col] = (__bf16)v;
      }
    }
  }
  __syncthreads();

  // layer 2: [32x256]@[256x128]+b2, relu -> sH2  (8 ni tiles, 1 per wave)
  {
    int ni = wave;
    const __bf16* bp = W2t + (size_t)(ni * 16 + nl) * 256 + half * 8;
    bf16x16 b[8];
#pragma unroll
    for (int kk = 0; kk < 8; ++kk) b[kk] = ld_b16(bp + kk * 32);
    float bias = b2[ni * 16 + nl];
    int col = ni * 16 + nl;
#pragma unroll
    for (int mi = 0; mi < 2; ++mi) {
      f32x8 acc = splat8(bias);
      const __bf16* ap = sH1 + (mi * 16 + nl) * 256 + half * 8;
#pragma unroll
      for (int kk = 0; kk < 8; ++kk)
        acc = wmma_bf16(ld_a16(ap + kk * 32), b[kk], acc);
#pragma unroll
      for (int j = 0; j < 8; ++j) {
        float v = acc[j] > 0.0f ? acc[j] : 0.0f;
        sH2[(mi * 16 + half * 8 + j) * 128 + col] = (__bf16)v;
      }
    }
  }
  __syncthreads();

  // layer 3: [32x128]@[128x128]+b3 -> out (last row forced to 0)
  {
    int ni = wave;
    const __bf16* bp = W3t + (size_t)(ni * 16 + nl) * 128 + half * 8;
    bf16x16 b[4];
#pragma unroll
    for (int kk = 0; kk < 4; ++kk) b[kk] = ld_b16(bp + kk * 32);
    float bias = b3[ni * 16 + nl];
    int col = ni * 16 + nl;
#pragma unroll
    for (int mi = 0; mi < 2; ++mi) {
      f32x8 acc = splat8(bias);
      const __bf16* ap = sH2 + (mi * 16 + nl) * 128 + half * 8;
#pragma unroll
      for (int kk = 0; kk < 4; ++kk)
        acc = wmma_bf16(ld_a16(ap + kk * 32), b[kk], acc);
      int gr0 = m0g + mi * 16 + half * 8;
#pragma unroll
      for (int j = 0; j < 8; ++j) {
        int gr = gr0 + j;
        float v = (gr == nRows - 1) ? 0.0f : acc[j];
        if (gr < nRows) out[(size_t)gr * 128 + col] = v;
      }
    }
  }
}

extern "C" void kernel_launch(void* const* d_in, const int* in_sizes, int n_in,
                              void* d_out, int out_size, void* d_ws, size_t ws_size,
                              hipStream_t stream) {
  const float* items      = (const float*)d_in[0];
  const int*   parents    = (const int*)d_in[1];
  const float* operations = (const float*)d_in[2];
  const int*   iep        = (const int*)d_in[3];
  const int*   iec        = (const int*)d_in[4];
  const int*   oei        = (const int*)d_in[5];
  const int*   oeo        = (const int*)d_in[6];
  const int N  = in_sizes[1];
  const int EI = in_sizes[3];
  const int EO = in_sizes[5];

  // params in dict order: self, parent, children, operations (W1,b1,W2,b2 each),
  // combined (W1,b1,W2,b2,W3,b3)
  const float* p[22];
  for (int i = 0; i < 22; ++i) p[i] = (const float*)d_in[7 + i];

  char* ws = (char*)d_ws;
  size_t off = 0;
  auto alloc = [&](size_t bytes) -> void* {
    void* r = ws + off;
    off += (bytes + 255) & ~(size_t)255;
    return r;
  };
  float* aggC = (float*)alloc((size_t)N * 128 * 4);
  float* aggO = (float*)alloc((size_t)N * 128 * 4);
  float* cat  = (float*)alloc((size_t)N * 512 * 4);
  __bf16* w1t[4]; __bf16* w2t[4];
  for (int br = 0; br < 4; ++br) {
    w1t[br] = (__bf16*)alloc((size_t)256 * 128 * 2);
    w2t[br] = (__bf16*)alloc((size_t)128 * 256 * 2);
  }
  __bf16* c1t = (__bf16*)alloc((size_t)256 * 512 * 2);
  __bf16* c2t = (__bf16*)alloc((size_t)128 * 256 * 2);
  __bf16* c3t = (__bf16*)alloc((size_t)128 * 128 * 2);

  hipMemsetAsync(aggC, 0, (size_t)N * 128 * 4, stream);
  hipMemsetAsync(aggO, 0, (size_t)N * 128 * 4, stream);

  auto tr = [&](const float* W, __bf16* Wt, int fin, int fout) {
    int tot = fin * fout;
    wtrans_kernel<<<(tot + 255) / 256, 256, 0, stream>>>(W, Wt, fin, fout);
  };
  tr(p[0],  w1t[0], 128, 256); tr(p[2],  w2t[0], 256, 128);  // self
  tr(p[4],  w1t[1], 128, 256); tr(p[6],  w2t[1], 256, 128);  // parent
  tr(p[8],  w1t[2], 128, 256); tr(p[10], w2t[2], 256, 128);  // children
  tr(p[12], w1t[3], 128, 256); tr(p[14], w2t[3], 256, 128);  // operations
  tr(p[16], c1t, 512, 256); tr(p[18], c2t, 256, 128); tr(p[20], c3t, 128, 128);

  scatter_add_kernel<<<(EI + 7) / 8, 256, 0, stream>>>(items, iep, iec, aggC, EI);
  scatter_add_kernel<<<(EO + 7) / 8, 256, 0, stream>>>(operations, oei, oeo, aggO, EO);

  int mb = (N + 63) / 64;
  // concat column layout: [parent | children | ops | self]
  mlp2_kernel<<<mb, 256, 0, stream>>>(items, parents, N, w1t[1], p[5],  w2t[1], p[7],  cat, 512, 0);
  mlp2_kernel<<<mb, 256, 0, stream>>>(aggC,  nullptr, N, w1t[2], p[9],  w2t[2], p[11], cat, 512, 128);
  mlp2_kernel<<<mb, 256, 0, stream>>>(aggO,  nullptr, N, w1t[3], p[13], w2t[3], p[15], cat, 512, 256);
  mlp2_kernel<<<mb, 256, 0, stream>>>(items, nullptr, N, w1t[0], p[1],  w2t[0], p[3],  cat, 512, 384);

  mlp3_kernel<<<(N + 31) / 32, 256, 0, stream>>>(cat, N, c1t, p[17], c2t, p[19],
                                                 c3t, p[21], (float*)d_out);
}
